// FutureChunkStageLabeler_64699387346992
// MI455X (gfx1250) — matrix operand order repaired and checked by reading, not compiled, not run on hardware
//
#include <hip/hip_runtime.h>
#include <math.h>

typedef __bf16 v16bf __attribute__((ext_vector_type(16)));
typedef __bf16 v8bf  __attribute__((ext_vector_type(8)));
typedef float  v8f   __attribute__((ext_vector_type(8)));

#define BN   4096
#define HN   64
#define DN   32
#define HIDN 256
#define KN   4
#define NPOS (BN*HN)          // 262144
#define NTOT ((size_t)BN*HN*DN) // 8388608

__device__ __forceinline__ float gelu_exact(float x) {
    return 0.5f * x * (1.0f + erff(x * 0.70710678118654752f));
}

// ---------------------------------------------------------------------------
// fp32 -> bf16 conversion
// ---------------------------------------------------------------------------
__global__ void f32_to_bf16_kernel(const float* __restrict__ in, __bf16* __restrict__ out, size_t n) {
    size_t i = (size_t)blockIdx.x * blockDim.x + threadIdx.x;
    if (i < n) out[i] = (__bf16)in[i];
}

// ---------------------------------------------------------------------------
// Weight packers -> CDNA5 WMMA B-matrix lane layout:
//   packed[((coTile*steps + s)*32 + lane)*16 + j]
//   lane: col = lane&15, half = lane>>4 ; element j -> K = s*32 + (j>>3)*16 + half*8 + (j&7)
// ---------------------------------------------------------------------------
__global__ void pack_linear_kernel(const float* __restrict__ W, __bf16* __restrict__ wp,
                                   int Kdim, int Nout, int NoutPad, size_t total) {
    size_t i = (size_t)blockIdx.x * blockDim.x + threadIdx.x;
    if (i >= total) return;
    int steps = Kdim >> 5;
    int j    = (int)(i & 15);
    int lane = (int)((i >> 4) & 31);
    size_t g = i >> 9;                  // coTile*steps + s
    int s      = (int)(g % steps);
    int coTile = (int)(g / steps);
    int half = lane >> 4, col = lane & 15;
    int kk = s * 32 + (j >> 3) * 16 + half * 8 + (j & 7);
    int co = coTile * 16 + col;
    float v = (co < Nout) ? W[(size_t)kk * Nout + co] : 0.0f;
    wp[i] = (__bf16)v;
}

// conv weights (Cout, Cin, KS=5); K ordering kk = k*Cin + ci
__global__ void pack_conv_kernel(const float* __restrict__ W, __bf16* __restrict__ wp,
                                 int Cin, int Cout, size_t total) {
    size_t i = (size_t)blockIdx.x * blockDim.x + threadIdx.x;
    if (i >= total) return;
    int steps = (Cin * 5) >> 5;
    int j    = (int)(i & 15);
    int lane = (int)((i >> 4) & 31);
    size_t g = i >> 9;
    int s      = (int)(g % steps);
    int coTile = (int)(g / steps);
    int half = lane >> 4, col = lane & 15;
    int kk = s * 32 + (j >> 3) * 16 + half * 8 + (j & 7);
    int k  = kk / Cin;
    int ci = kk % Cin;
    int co = coTile * 16 + col;
    wp[i] = (__bf16)W[((size_t)co * Cin + ci) * 5 + k];
}

// ---------------------------------------------------------------------------
// Implicit-im2col conv1d GEMM (same padding, KS=5) + bias + GELU, bf16 out.
// M = positions (16/tile), N = out channels, K = Cin*5 (bf16 WMMA, K-step 32).
// Each wave: 1 position tile x NB channel tiles (A fragment reused NB times).
// ---------------------------------------------------------------------------
template <int CIN, int NB>
__global__ __launch_bounds__(256)
void conv_gemm_kernel(const __bf16* __restrict__ actIn,   // [NPOS, CIN]
                      const __bf16* __restrict__ wpack,
                      const float*  __restrict__ bias,
                      __bf16*       __restrict__ actOut,  // [NPOS, COUT]
                      int Npos, int COUT) {
    const int wid  = (int)((blockIdx.x * blockDim.x + threadIdx.x) >> 5);
    const int lane = threadIdx.x & 31;
    const int coTiles  = COUT >> 4;
    const int coGroups = coTiles / NB;
    const int nTiles   = Npos >> 4;
    if (wid >= nTiles * coGroups) return;
    const int nTile = wid / coGroups;
    const int coG   = wid % coGroups;
    const int n0    = nTile * 16;
    const int half  = lane >> 4, lm = lane & 15;
    const int rowA  = n0 + lm;          // A-matrix row (M = lm)
    const int h     = rowA & (HN - 1);
    constexpr int STEPS = CIN * 5 / 32;
    const size_t tileStride = (size_t)STEPS * 512;

    v8f zero = {0.f,0.f,0.f,0.f,0.f,0.f,0.f,0.f};
    v8f acc[NB];
#pragma unroll
    for (int t = 0; t < NB; ++t) acc[t] = zero;

    for (int s = 0; s < STEPS; ++s) {
        int k, cibase;
        if constexpr (CIN == 32) { k = s;      cibase = 0; }
        else                     { k = s >> 3; cibase = (s & 7) * 32; }
        const int hp = h + k - 2;
        v16bf a;
        if (hp >= 0 && hp < HN) {
            const __bf16* ap = actIn + (size_t)(rowA + (k - 2)) * CIN + cibase;
            v8bf lo = *(const v8bf*)(ap + half * 8);        // K = half*8 + 0..7
            v8bf hi = *(const v8bf*)(ap + 16 + half * 8);   // K = 16 + half*8 + 0..7
#pragma unroll
            for (int j = 0; j < 8; ++j) { a[j] = lo[j]; a[8 + j] = hi[j]; }
        } else {
#pragma unroll
            for (int j = 0; j < 16; ++j) a[j] = (__bf16)0.0f;
        }
#pragma unroll
        for (int t = 0; t < NB; ++t) {
            const __bf16* wp = wpack + (size_t)(coG * NB + t) * tileStride
                                     + (size_t)s * 512 + lane * 16;
            v8bf b0 = *(const v8bf*)(wp);
            v8bf b1 = *(const v8bf*)(wp + 8);
            v16bf bm;
#pragma unroll
            for (int j = 0; j < 8; ++j) { bm[j] = b0[j]; bm[8 + j] = b1[j]; }
            acc[t] = __builtin_amdgcn_wmma_f32_16x16x32_bf16(
                false, a, false, bm, (short)0, acc[t], false, false);
        }
    }
#pragma unroll
    for (int t = 0; t < NB; ++t) {
        const int co = (coG * NB + t) * 16 + lm;   // D col = lane&15
        const float bv = bias[co];
#pragma unroll
        for (int r = 0; r < 8; ++r) {              // D row = half*8 + r
            const int rowD = n0 + half * 8 + r;
            float v = acc[t][r] + bv;
            v = gelu_exact(v);
            actOut[(size_t)rowD * COUT + co] = (__bf16)v;
        }
    }
}

// ---------------------------------------------------------------------------
// Generic linear GEMM (M rows x Kdim) @ (Kdim x Nout) + bias [+ GELU]
// ---------------------------------------------------------------------------
template <bool GELU_OUT, bool OUT_BF16>
__global__ __launch_bounds__(256)
void linear_gemm_kernel(const __bf16* __restrict__ A,       // [M, Kdim]
                        const __bf16* __restrict__ wpack,
                        const float*  __restrict__ bias,
                        float*        __restrict__ outF,
                        __bf16*       __restrict__ outB,
                        int Kdim, int Mrows, int Nout, int NoutPad) {
    const int wid  = (int)((blockIdx.x * blockDim.x + threadIdx.x) >> 5);
    const int lane = threadIdx.x & 31;
    const int coTiles = NoutPad >> 4;
    const int mTiles  = Mrows >> 4;
    if (wid >= mTiles * coTiles) return;
    const int mTile  = wid / coTiles;
    const int coTile = wid % coTiles;
    const int half = lane >> 4, lm = lane & 15;
    const int row = mTile * 16 + lm;
    const int steps = Kdim >> 5;

    v8f acc = {0.f,0.f,0.f,0.f,0.f,0.f,0.f,0.f};
    const __bf16* ap0 = A + (size_t)row * Kdim;
    const __bf16* wp  = wpack + (size_t)coTile * steps * 512 + lane * 16;
    for (int s = 0; s < steps; ++s) {
        v16bf a, bm;
        v8bf alo = *(const v8bf*)(ap0 + s * 32 + half * 8);
        v8bf ahi = *(const v8bf*)(ap0 + s * 32 + 16 + half * 8);
        v8bf b0  = *(const v8bf*)(wp);
        v8bf b1  = *(const v8bf*)(wp + 8);
#pragma unroll
        for (int j = 0; j < 8; ++j) {
            a[j] = alo[j]; a[8 + j] = ahi[j];
            bm[j] = b0[j]; bm[8 + j] = b1[j];
        }
        wp += 512;
        acc = __builtin_amdgcn_wmma_f32_16x16x32_bf16(
            false, a, false, bm, (short)0, acc, false, false);
    }
    const int co = coTile * 16 + lm;
    if (co < Nout) {
        const float bv = bias[co];
#pragma unroll
        for (int r = 0; r < 8; ++r) {
            const int rowD = mTile * 16 + half * 8 + r;
            float v = acc[r] + bv;
            if (GELU_OUT) v = gelu_exact(v);
            if (OUT_BF16) outB[(size_t)rowD * Nout + co] = (__bf16)v;
            else          outF[(size_t)rowD * Nout + co] = v;
        }
    }
}

// ---------------------------------------------------------------------------
// Adaptive average pool over H (gelu already applied by conv epilogue)
// ---------------------------------------------------------------------------
__global__ void pool_kernel(const __bf16* __restrict__ act, __bf16* __restrict__ pooled) {
    int i = blockIdx.x * blockDim.x + threadIdx.x;
    if (i >= BN * HIDN) return;
    int b = i >> 8, c = i & 255;
    const __bf16* p = act + ((size_t)b * HN) * HIDN + c;
    float s = 0.f;
#pragma unroll 8
    for (int h = 0; h < HN; ++h) s += (float)p[(size_t)h * HIDN];
    pooled[i] = (__bf16)(s * (1.0f / HN));
}

// ---------------------------------------------------------------------------
// Zero loss accumulators
// ---------------------------------------------------------------------------
__global__ void zero_accum_kernel(float* a) {
    if (threadIdx.x < 16) a[threadIdx.x] = 0.0f;
}

// ---------------------------------------------------------------------------
// Per-batch head: softmax q / entropy / balance sums, sigmoid+sort positions,
// endpoint clamp, spacing penalty.  accum: [0]sse [1]l1 [2]ent [3]q0 [4]q1 [5]sp
// ---------------------------------------------------------------------------
__global__ void batch_head_kernel(const float* __restrict__ rawpos,
                                  const float* __restrict__ mixlog,
                                  float* __restrict__ pos4, int* __restrict__ ord4,
                                  float* __restrict__ q1g, float* __restrict__ accum) {
    int b = blockIdx.x * blockDim.x + threadIdx.x;
    if (b >= BN) return;
    // softmax over 2 logits
    float l0 = mixlog[b * 2 + 0], l1 = mixlog[b * 2 + 1];
    float m = fmaxf(l0, l1);
    float e0 = expf(l0 - m), e1 = expf(l1 - m);
    float inv = 1.0f / (e0 + e1);
    float q0 = e0 * inv, q1 = e1 * inv;
    q1g[b] = q1;
    float entc = q0 * logf(q0 + 1e-8f) + q1 * logf(q1 + 1e-8f);
    // positions: sigmoid * (H-1), stable argsort of 4
    float sc[KN]; int id[KN];
#pragma unroll
    for (int i = 0; i < KN; ++i) {
        float rp = rawpos[b * KN + i];
        sc[i] = 63.0f / (1.0f + expf(-rp));
        id[i] = i;
    }
#pragma unroll
    for (int i = 1; i < KN; ++i) {
        float v = sc[i]; int vi = id[i]; int j = i - 1;
        while (j >= 0 && sc[j] > v) { sc[j + 1] = sc[j]; id[j + 1] = id[j]; --j; }
        sc[j + 1] = v; id[j + 1] = vi;
    }
    sc[0] = 0.0f; sc[KN - 1] = 63.0f;
    float sp = 0.f;
#pragma unroll
    for (int i = 0; i < KN - 1; ++i) {
        float g = sc[i + 1] - sc[i];
        float r = fmaxf(10.5f - g, 0.0f);   // min_gap = 0.5*63/3
        sp += r * r;
    }
    sp *= (1.0f / (KN - 1));
#pragma unroll
    for (int i = 0; i < KN; ++i) { pos4[b * KN + i] = sc[i]; ord4[b * KN + i] = id[i]; }
    atomicAdd(&accum[2], entc);
    atomicAdd(&accum[3], q0);
    atomicAdd(&accum[4], q1);
    atomicAdd(&accum[5], sp);
}

// ---------------------------------------------------------------------------
// Interp + recon/L1 loss partial sums. recon = coarse + q1*residual.
// One thread per (b,h), looping D=32.
// ---------------------------------------------------------------------------
__global__ __launch_bounds__(256)
void recon_loss_kernel(const float* __restrict__ x,
                       const float* __restrict__ anchor,    // [B, K*D]
                       const float* __restrict__ residual,  // [B, H*D]
                       const float* __restrict__ pos4,
                       const int*   __restrict__ ord4,
                       const float* __restrict__ q1g,
                       float* __restrict__ accum) {
    int bh = blockIdx.x * blockDim.x + threadIdx.x;
    float lsse = 0.f, ll1 = 0.f;
    if (bh < BN * HN) {
        int b = bh >> 6, h = bh & (HN - 1);
        float p0 = pos4[b * 4 + 0], p1 = pos4[b * 4 + 1];
        float p2 = pos4[b * 4 + 2], p3 = pos4[b * 4 + 3];
        float t = (float)h;
        int idx = (p0 <= t) + (p1 <= t) + (p2 <= t) + (p3 <= t) - 1;
        idx = min(max(idx, 0), 2);
        float left  = (idx == 0) ? p0 : ((idx == 1) ? p1 : p2);
        float right = (idx == 0) ? p1 : ((idx == 1) ? p2 : p3);
        float w = (t - left) / (right - left + 1e-8f);
        w = fminf(fmaxf(w, 0.0f), 1.0f);
        int ol = ord4[b * 4 + idx], orr = ord4[b * 4 + idx + 1];
        const float* vl = anchor + (size_t)b * (KN * DN) + ol  * DN;
        const float* vr = anchor + (size_t)b * (KN * DN) + orr * DN;
        const float* rs = residual + (size_t)bh * DN;
        const float* xx = x + (size_t)bh * DN;
        float qq = q1g[b];
#pragma unroll 8
        for (int d = 0; d < DN; ++d) {
            float c = vl[d] + w * (vr[d] - vl[d]);
            float r = rs[d];
            float diff = c + qq * r - xx[d];
            lsse += diff * diff;
            ll1  += fabsf(r);
        }
    }
    __shared__ float s0[256], s1[256];
    int tid = threadIdx.x;
    s0[tid] = lsse; s1[tid] = ll1;
    __syncthreads();
    for (int st = 128; st > 0; st >>= 1) {
        if (tid < st) { s0[tid] += s0[tid + st]; s1[tid] += s1[tid + st]; }
        __syncthreads();
    }
    if (tid == 0) { atomicAdd(&accum[0], s0[0]); atomicAdd(&accum[1], s1[0]); }
}

__global__ void final_kernel(const float* __restrict__ accum, float* __restrict__ out) {
    if (threadIdx.x == 0 && blockIdx.x == 0) {
        const float N = (float)NTOT, Bf = (float)BN;
        float recon = accum[0] / N;
        float l1    = accum[1] / N;
        float ent   = -accum[2] / Bf;
        float mq0 = accum[3] / Bf, mq1 = accum[4] / Bf;
        float bal = (mq0 - 0.5f) * (mq0 - 0.5f) + (mq1 - 0.5f) * (mq1 - 0.5f);
        float sp  = accum[5] / Bf;
        out[0] = recon + 0.05f * l1 + 0.01f * ent + 0.1f * bal + 0.1f * sp;
    }
}

// ---------------------------------------------------------------------------
extern "C" void kernel_launch(void* const* d_in, const int* in_sizes, int n_in,
                              void* d_out, int out_size, void* d_ws, size_t ws_size,
                              hipStream_t stream) {
    const float* x       = (const float*)d_in[0];
    const float* conv_w0 = (const float*)d_in[1];
    const float* conv_b0 = (const float*)d_in[2];
    const float* conv_w1 = (const float*)d_in[3];
    const float* conv_b1 = (const float*)d_in[4];
    const float* conv_w2 = (const float*)d_in[5];
    const float* conv_b2 = (const float*)d_in[6];
    const float* proj_w  = (const float*)d_in[7];
    const float* proj_b  = (const float*)d_in[8];
    const float* val_w1  = (const float*)d_in[9];
    const float* val_b1  = (const float*)d_in[10];
    const float* val_w2  = (const float*)d_in[11];
    const float* val_b2  = (const float*)d_in[12];
    const float* pos_w1  = (const float*)d_in[13];
    const float* pos_b1  = (const float*)d_in[14];
    const float* pos_w2  = (const float*)d_in[15];
    const float* pos_b2  = (const float*)d_in[16];
    const float* res_w1  = (const float*)d_in[17];
    const float* res_b1  = (const float*)d_in[18];
    const float* res_w2  = (const float*)d_in[19];
    const float* res_b2  = (const float*)d_in[20];
    const float* mix_w   = (const float*)d_in[21];
    const float* mix_b   = (const float*)d_in[22];
    float* out = (float*)d_out;

    // -------- workspace carve-up (256B aligned) --------
    size_t off = 0;
    auto alloc = [&](size_t bytes) -> void* {
        void* p = (char*)d_ws + off;
        off += (bytes + 255) & ~(size_t)255;
        return p;
    };
    float*  accum   = (float*)alloc(64 * 4);
    __bf16* act0    = (__bf16*)alloc((size_t)NPOS * DN   * 2);   // 16 MB
    __bf16* actA    = (__bf16*)alloc((size_t)NPOS * HIDN * 2);   // 128 MB
    __bf16* actB    = (__bf16*)alloc((size_t)NPOS * HIDN * 2);   // 128 MB
    __bf16* pk_c0   = (__bf16*)alloc((size_t)16 * 5  * 512 * 2);
    __bf16* pk_c1   = (__bf16*)alloc((size_t)16 * 40 * 512 * 2);
    __bf16* pk_c2   = (__bf16*)alloc((size_t)16 * 40 * 512 * 2);
    __bf16* pk_proj = (__bf16*)alloc((size_t)16 * 8 * 512 * 2);
    __bf16* pk_v1   = (__bf16*)alloc((size_t)16 * 8 * 512 * 2);
    __bf16* pk_v2   = (__bf16*)alloc((size_t)8  * 8 * 512 * 2);
    __bf16* pk_p1   = (__bf16*)alloc((size_t)16 * 8 * 512 * 2);
    __bf16* pk_p2   = (__bf16*)alloc((size_t)1  * 8 * 512 * 2);
    __bf16* pk_r1   = (__bf16*)alloc((size_t)16 * 8 * 512 * 2);
    __bf16* pk_r2   = (__bf16*)alloc((size_t)128 * 8 * 512 * 2);
    __bf16* pk_mix  = (__bf16*)alloc((size_t)1  * 8 * 512 * 2);
    __bf16* pooled  = (__bf16*)alloc((size_t)BN * HIDN * 2);
    __bf16* enc     = (__bf16*)alloc((size_t)BN * HIDN * 2);
    __bf16* vh      = (__bf16*)alloc((size_t)BN * HIDN * 2);
    __bf16* ph      = (__bf16*)alloc((size_t)BN * HIDN * 2);
    __bf16* rh      = (__bf16*)alloc((size_t)BN * HIDN * 2);
    float*  anchor  = (float*)alloc((size_t)BN * KN * DN * 4);
    float*  rawpos  = (float*)alloc((size_t)BN * KN * 4);
    float*  mixlog  = (float*)alloc((size_t)BN * 2 * 4);
    float*  residual= (float*)alloc((size_t)BN * HN * DN * 4);   // 32 MB
    float*  pos4    = (float*)alloc((size_t)BN * KN * 4);
    int*    ord4    = (int*)alloc((size_t)BN * KN * 4);
    float*  q1g     = (float*)alloc((size_t)BN * 4);

    auto cdiv = [](size_t a, size_t b) { return (unsigned)((a + b - 1) / b); };

    // -------- prep --------
    zero_accum_kernel<<<1, 64, 0, stream>>>(accum);
    f32_to_bf16_kernel<<<cdiv(NTOT, 256), 256, 0, stream>>>(x, act0, NTOT);

    { size_t t = (size_t)16 * 5 * 512;
      pack_conv_kernel<<<cdiv(t, 256), 256, 0, stream>>>(conv_w0, pk_c0, 32, 256, t); }
    { size_t t = (size_t)16 * 40 * 512;
      pack_conv_kernel<<<cdiv(t, 256), 256, 0, stream>>>(conv_w1, pk_c1, 256, 256, t);
      pack_conv_kernel<<<cdiv(t, 256), 256, 0, stream>>>(conv_w2, pk_c2, 256, 256, t); }
    { size_t t = (size_t)16 * 8 * 512;
      pack_linear_kernel<<<cdiv(t, 256), 256, 0, stream>>>(proj_w, pk_proj, 256, 256, 256, t);
      pack_linear_kernel<<<cdiv(t, 256), 256, 0, stream>>>(val_w1, pk_v1, 256, 256, 256, t);
      pack_linear_kernel<<<cdiv(t, 256), 256, 0, stream>>>(pos_w1, pk_p1, 256, 256, 256, t);
      pack_linear_kernel<<<cdiv(t, 256), 256, 0, stream>>>(res_w1, pk_r1, 256, 256, 256, t); }
    { size_t t = (size_t)8 * 8 * 512;
      pack_linear_kernel<<<cdiv(t, 256), 256, 0, stream>>>(val_w2, pk_v2, 256, 128, 128, t); }
    { size_t t = (size_t)1 * 8 * 512;
      pack_linear_kernel<<<cdiv(t, 256), 256, 0, stream>>>(pos_w2, pk_p2, 256, 4, 16, t);
      pack_linear_kernel<<<cdiv(t, 256), 256, 0, stream>>>(mix_w,  pk_mix, 256, 2, 16, t); }
    { size_t t = (size_t)128 * 8 * 512;
      pack_linear_kernel<<<cdiv(t, 256), 256, 0, stream>>>(res_w2, pk_r2, 256, 2048, 2048, t); }

    // -------- conv stack (WMMA implicit-im2col GEMMs) --------
    {   // waves = (NPOS/16) * (coTiles/NB) = 16384 * 4
        size_t waves = (size_t)(NPOS / 16) * (256 / 16 / 4);
        unsigned blocks = cdiv(waves * 32, 256);
        conv_gemm_kernel<32, 4><<<blocks, 256, 0, stream>>>(act0, pk_c0, conv_b0, actA, NPOS, 256);
        conv_gemm_kernel<256, 4><<<blocks, 256, 0, stream>>>(actA, pk_c1, conv_b1, actB, NPOS, 256);
        conv_gemm_kernel<256, 4><<<blocks, 256, 0, stream>>>(actB, pk_c2, conv_b2, actA, NPOS, 256);
    }

    // -------- pool + encoder --------
    pool_kernel<<<cdiv((size_t)BN * HIDN, 256), 256, 0, stream>>>(actA, pooled);

    auto lin_grid = [&](int coTilesPad) { return cdiv((size_t)(BN / 16) * coTilesPad * 32, 256); };
    linear_gemm_kernel<true,  true ><<<lin_grid(16), 256, 0, stream>>>(pooled, pk_proj, proj_b, nullptr, enc, 256, BN, 256, 256);
    // -------- heads --------
    linear_gemm_kernel<true,  true ><<<lin_grid(16), 256, 0, stream>>>(enc, pk_v1, val_b1, nullptr, vh, 256, BN, 256, 256);
    linear_gemm_kernel<false, false><<<lin_grid(8),  256, 0, stream>>>(vh,  pk_v2, val_b2, anchor, nullptr, 256, BN, 128, 128);
    linear_gemm_kernel<true,  true ><<<lin_grid(16), 256, 0, stream>>>(enc, pk_p1, pos_b1, nullptr, ph, 256, BN, 256, 256);
    linear_gemm_kernel<false, false><<<lin_grid(1),  256, 0, stream>>>(ph,  pk_p2, pos_b2, rawpos, nullptr, 256, BN, 4, 16);
    linear_gemm_kernel<true,  true ><<<lin_grid(16), 256, 0, stream>>>(enc, pk_r1, res_b1, nullptr, rh, 256, BN, 256, 256);
    linear_gemm_kernel<false, false><<<lin_grid(128),256, 0, stream>>>(rh,  pk_r2, res_b2, residual, nullptr, 256, BN, 2048, 2048);
    linear_gemm_kernel<false, false><<<lin_grid(1),  256, 0, stream>>>(enc, pk_mix, mix_b, mixlog, nullptr, 256, BN, 2, 16);

    // -------- scalar tail + losses --------
    batch_head_kernel<<<cdiv(BN, 256), 256, 0, stream>>>(rawpos, mixlog, pos4, ord4, q1g, accum);
    recon_loss_kernel<<<cdiv((size_t)BN * HN, 256), 256, 0, stream>>>(x, anchor, residual, pos4, ord4, q1g, accum);
    final_kernel<<<1, 1, 0, stream>>>(accum, out);
}
